// PrefixGraphTransformerPLUS_24043226923200
// MI455X (gfx1250) — matrix operand order repaired and checked by reading, not compiled, not run on hardware
//
#include <hip/hip_runtime.h>
#include <hip/hip_bf16.h>

typedef _Float16 half_t;
typedef __attribute__((ext_vector_type(16))) _Float16 v16h;
typedef __attribute__((ext_vector_type(8)))  float    v8f;

#define NB    8
#define NA_   1024
#define KNB   32
#define CF    140
#define CPF   20
#define DA    128
#define DP    64
#define NH    8
#define DHD   16
#define DOP   32
#define NLAY  3
#define ROWS_A (NB * NA_)        /* 8192   */
#define ROWS_Z (ROWS_A * KNB)    /* 262144 */
#define EPS_  1e-5f

// ---------------------------------------------------------------------------
// WMMA fragment loaders (16x16x32 f16, wave32; contiguous k order used by this code)
// A: 16x32 f16, lane l: row m=l&15, half h=l>>4; VGPR v pairs K =
//    h*16 + 2v
// B: 32x16 f16 stored transposed Bt[N][Kpad]; lane l: col n=l&15, h=l>>4;
//    VGPR v pairs K = h*16 + 2v  (contiguous in Bt row -> dword loads)
// C/D: f32, lane l: col n=l&15; VGPR r: row m = (l>>4)*8 + r
// ---------------------------------------------------------------------------
__device__ __forceinline__ v16h load_a_frag(const half_t* A, int ldk, int kb, int lane) {
  int m = lane & 15;
  int h = lane >> 4;
  const half_t* p = A + m * ldk + kb;
  v16h o;
#pragma unroll
  for (int v = 0; v < 8; ++v) {
    int k = h * 16 + (v << 1);
    o[2 * v]     = p[k];
    o[2 * v + 1] = p[k + 1];
  }
  return o;
}

__device__ __forceinline__ v16h load_b_frag(const half_t* Bt, int ldk, int kb, int lane) {
  int n = lane & 15;
  int h = lane >> 4;
  const half_t* p = Bt + (size_t)n * ldk + kb + h * 16;
  v16h o;
#pragma unroll
  for (int v = 0; v < 8; ++v) {
    o[2 * v]     = p[2 * v];
    o[2 * v + 1] = p[2 * v + 1];
  }
  return o;
}

__device__ __forceinline__ v8f wmma_f16(v16h a, v16h b, v8f c) {
  return __builtin_amdgcn_wmma_f32_16x16x32_f16(false, a, false, b, (short)0, c, false, false);
}

__device__ __forceinline__ float wave_sum32(float v) {
#pragma unroll
  for (int off = 16; off > 0; off >>= 1) v += __shfl_xor(v, off, 32);
  return v;
}

// ---------------------------------------------------------------------------
// Weight prep: fp32 [K][N] -> f16 transposed, zero-padded [N][KPAD]
// ---------------------------------------------------------------------------
__global__ void convert_wt_kernel(const float* __restrict__ src, half_t* __restrict__ dst,
                                  int K, int N, int KPAD) {
  int idx = blockIdx.x * blockDim.x + threadIdx.x;
  if (idx >= N * KPAD) return;
  int n = idx / KPAD, k = idx % KPAD;
  dst[idx] = (k < K) ? (half_t)src[(size_t)k * N + n] : (half_t)0.f;
}

__global__ void fill0_kernel(float* __restrict__ p, size_t n) {
  size_t i = (size_t)blockIdx.x * blockDim.x + threadIdx.x;
  if (i < n) p[i] = 0.f;
}

// ---------------------------------------------------------------------------
// Generic row GEMM: Y[rows][N] (=|+=) X[rows][KDIM] @ W[KDIM][N] + bias
// Block = 128 threads = 4 waves; each wave owns a 16-row tile.
// X rows converted fp32->f16 into LDS; W fragments stream from f16 Wt in L2.
// ---------------------------------------------------------------------------
template <int KDIM, int KPAD, int N>
__global__ __launch_bounds__(128) void gemm_rows_kernel(
    const float* __restrict__ X, const half_t* __restrict__ Wt,
    const float* __restrict__ bias, float* __restrict__ Y, int mode /*0=store,1=add*/) {
  __shared__ half_t ldsx[4 * 16 * KPAD];
  int lane = threadIdx.x & 31, wave = threadIdx.x >> 5;
  int row0 = (blockIdx.x * 4 + wave) * 16;
  half_t* mx = ldsx + wave * 16 * KPAD;
  for (int j = lane; j < 16 * KPAD; j += 32) {
    int r = j / KPAD, c = j % KPAD;
    float v = (c < KDIM) ? X[(size_t)(row0 + r) * KDIM + c] : 0.f;
    mx[j] = (half_t)v;
  }
  __syncthreads();
  int n15 = lane & 15, h = lane >> 4;
#pragma unroll
  for (int nt = 0; nt < N / 16; ++nt) {
    int ncol = nt * 16;
    float bv = bias ? bias[ncol + n15] : 0.f;
    v8f c;
#pragma unroll
    for (int r = 0; r < 8; ++r) c[r] = bv;
#pragma unroll
    for (int kb = 0; kb < KPAD; kb += 32) {
      v16h af = load_a_frag(mx, KPAD, kb, lane);
      v16h bf = load_b_frag(Wt + (size_t)ncol * KPAD, KPAD, kb, lane);
      c = wmma_f16(af, bf, c);
    }
#pragma unroll
    for (int r = 0; r < 8; ++r) {
      int m = h * 8 + r;
      size_t off = (size_t)(row0 + m) * N + ncol + n15;
      if (mode) Y[off] += c[r]; else Y[off] = c[r];
    }
  }
}

// ---------------------------------------------------------------------------
// Fused 2-layer MLP: Y = relu(X@W1+b1)@W2+b2 ; hidden kept in LDS as f16
// ---------------------------------------------------------------------------
template <int K1, int KP1, int N1, int N2>
__global__ __launch_bounds__(128) void mlp_kernel(
    const float* __restrict__ X, const half_t* __restrict__ W1t, const float* __restrict__ b1,
    const half_t* __restrict__ W2t, const float* __restrict__ b2, float* __restrict__ Y) {
  __shared__ half_t ldsx[4 * 16 * KP1];
  __shared__ half_t ldsh[4 * 16 * N1];
  int lane = threadIdx.x & 31, wave = threadIdx.x >> 5;
  int row0 = (blockIdx.x * 4 + wave) * 16;
  half_t* mx = ldsx + wave * 16 * KP1;
  half_t* mh = ldsh + wave * 16 * N1;
  for (int j = lane; j < 16 * KP1; j += 32) {
    int r = j / KP1, c = j % KP1;
    float v = (c < K1) ? X[(size_t)(row0 + r) * K1 + c] : 0.f;
    mx[j] = (half_t)v;
  }
  __syncthreads();
  int n15 = lane & 15, h = lane >> 4;
  // GEMM1 + ReLU -> LDS hidden (f16)
#pragma unroll
  for (int nt = 0; nt < N1 / 16; ++nt) {
    int ncol = nt * 16;
    float bv = b1[ncol + n15];
    v8f c;
#pragma unroll
    for (int r = 0; r < 8; ++r) c[r] = bv;
#pragma unroll
    for (int kb = 0; kb < KP1; kb += 32) {
      v16h af = load_a_frag(mx, KP1, kb, lane);
      v16h bf = load_b_frag(W1t + (size_t)ncol * KP1, KP1, kb, lane);
      c = wmma_f16(af, bf, c);
    }
#pragma unroll
    for (int r = 0; r < 8; ++r) {
      int m = h * 8 + r;
      mh[m * N1 + ncol + n15] = (half_t)fmaxf(c[r], 0.f);
    }
  }
  __syncthreads();
  // GEMM2 -> Y
#pragma unroll
  for (int nt = 0; nt < N2 / 16; ++nt) {
    int ncol = nt * 16;
    float bv = b2[ncol + n15];
    v8f c;
#pragma unroll
    for (int r = 0; r < 8; ++r) c[r] = bv;
#pragma unroll
    for (int kb = 0; kb < N1; kb += 32) {
      v16h af = load_a_frag(mh, N1, kb, lane);
      v16h bf = load_b_frag(W2t + (size_t)ncol * N1, N1, kb, lane);
      c = wmma_f16(af, bf, c);
    }
#pragma unroll
    for (int r = 0; r < 8; ++r) {
      int m = h * 8 + r;
      Y[(size_t)(row0 + m) * N2 + ncol + n15] = c[r];
    }
  }
}

// ---------------------------------------------------------------------------
// Outer-product pair update: dZ[b,i,k,:] = (left[b,i]*right[b,nl]) @ Wp + bp
// (gather-commuted: right rows are pre-projected, gathered here, masked)
// ---------------------------------------------------------------------------
__global__ __launch_bounds__(128) void outer_product_kernel(
    const float* __restrict__ leftb, const float* __restrict__ rightp,
    const float* __restrict__ amask, const int* __restrict__ nl,
    const half_t* __restrict__ Wpt, const float* __restrict__ bp,
    float* __restrict__ dZ) {
  __shared__ half_t ldsx[4 * 16 * DOP];
  int lane = threadIdx.x & 31, wave = threadIdx.x >> 5;
  int row0 = (blockIdx.x * 4 + wave) * 16;
  half_t* mx = ldsx + wave * 16 * DOP;
  for (int j = lane; j < 16 * DOP; j += 32) {
    int r = j / DOP, c = j % DOP;
    int gr = row0 + r;
    int b = gr / (NA_ * KNB);
    int rem = gr % (NA_ * KNB);
    int i = rem / KNB;
    int g = nl[gr];
    float lv = leftb[(size_t)(b * NA_ + i) * DOP + c] * amask[b * NA_ + i];
    float rv = rightp[(size_t)(b * NA_ + g) * DOP + c] * amask[b * NA_ + g];
    mx[j] = (half_t)(lv * rv);
  }
  __syncthreads();
  int n15 = lane & 15, h = lane >> 4;
  v16h af = load_a_frag(mx, DOP, 0, lane);
#pragma unroll
  for (int nt = 0; nt < DP / 16; ++nt) {
    int ncol = nt * 16;
    float bv = bp[ncol + n15];
    v8f c;
#pragma unroll
    for (int r = 0; r < 8; ++r) c[r] = bv;
    v16h bf = load_b_frag(Wpt + (size_t)ncol * DOP, DOP, 0, lane);
    c = wmma_f16(af, bf, c);
#pragma unroll
    for (int r = 0; r < 8; ++r) {
      int m = h * 8 + r;
      dZ[(size_t)(row0 + m) * DP + ncol + n15] = c[r];
    }
  }
}

// ---------------------------------------------------------------------------
// Fused neighbor attention: one block per (b,i). Gathers projected k/v rows
// into LDS, logits (dot/4 + pair bias), masked softmax over K=32, weighted
// sum -> o[b,i,128]. ~0.1 GF/layer total: VALU kernel.
// ---------------------------------------------------------------------------
__global__ __launch_bounds__(128) void attention_kernel(
    const float* __restrict__ qp, const float* __restrict__ kp, const float* __restrict__ vp,
    const float* __restrict__ z, const float* __restrict__ Wb,
    const float* __restrict__ amask, const int* __restrict__ nl,
    float* __restrict__ o) {
  __shared__ float sk[KNB * DA];
  __shared__ float sv[KNB * DA];
  __shared__ float sq[DA];
  __shared__ float sl[NH * KNB];
  __shared__ int snl[KNB];
  __shared__ float sm[KNB];
  int tid = threadIdx.x;
  int bi = blockIdx.x;   // b*A + i
  int b = bi / NA_;
  size_t zbase = (size_t)bi * KNB * DP;
  if (tid < DA) sq[tid] = qp[(size_t)bi * DA + tid];
  if (tid < KNB) {
    int g = nl[(size_t)bi * KNB + tid];
    snl[tid] = g;
    sm[tid] = amask[b * NA_ + g];
  }
  __syncthreads();
  for (int t = tid; t < KNB * DA; t += 128) {
    int r = t / DA, c = t % DA;
    size_t gro = (size_t)(b * NA_ + snl[r]) * DA + c;
    sk[t] = kp[gro];
    sv[t] = vp[gro];
  }
  __syncthreads();
  for (int t = tid; t < NH * KNB; t += 128) {
    int hh = t / KNB, k = t % KNB;
    float dot = 0.f;
#pragma unroll
    for (int e = 0; e < DHD; ++e) dot += sq[hh * DHD + e] * sk[k * DA + hh * DHD + e];
    float bias = 0.f;
    const float* zr = z + zbase + (size_t)k * DP;
#pragma unroll 4
    for (int c = 0; c < DP; ++c) bias += zr[c] * Wb[c * NH + hh];
    float lg = dot * 0.25f + bias;   // 1/sqrt(DH)=0.25
    if (sm[k] <= 0.f) lg = -1e9f;
    sl[t] = lg;
  }
  __syncthreads();
  if (tid < NH) {
    float mx = -1e30f;
    for (int k = 0; k < KNB; ++k) mx = fmaxf(mx, sl[tid * KNB + k]);
    float s = 0.f;
    for (int k = 0; k < KNB; ++k) {
      float e = __expf(sl[tid * KNB + k] - mx);
      sl[tid * KNB + k] = e;
      s += e;
    }
    float inv = 1.f / s;
    for (int k = 0; k < KNB; ++k) sl[tid * KNB + k] *= inv;
  }
  __syncthreads();
  {
    int hh = tid / DHD, e = tid % DHD;
    float acc = 0.f;
#pragma unroll 4
    for (int k = 0; k < KNB; ++k) acc += sl[hh * KNB + k] * sv[k * DA + hh * DHD + e];
    o[(size_t)bi * DA + tid] = acc;
  }
}

// ---------------------------------------------------------------------------
// ResiDual step: x = LN(x+d)*g+b ; acc += d ; optional row mask on both.
// One wave per row (wave32), D/32 elems per lane, shfl reductions.
// ---------------------------------------------------------------------------
template <int D, bool HAS_MASK>
__global__ __launch_bounds__(256) void residual_ln_kernel(
    float* __restrict__ x, const float* __restrict__ d, float* __restrict__ acc,
    const float* __restrict__ g, const float* __restrict__ beta,
    const float* __restrict__ mask, int nrows) {
  constexpr int E = D / 32;
  int lane = threadIdx.x & 31;
  int row = blockIdx.x * (blockDim.x >> 5) + (threadIdx.x >> 5);
  if (row >= nrows) return;
  size_t base = (size_t)row * D + lane * E;
  float xv[E], av[E];
  float s = 0.f, s2 = 0.f;
#pragma unroll
  for (int e = 0; e < E; ++e) {
    float dv = d[base + e];
    float v = x[base + e] + dv;
    av[e] = acc[base + e] + dv;
    xv[e] = v;
    s += v;
    s2 += v * v;
  }
  s = wave_sum32(s);
  s2 = wave_sum32(s2);
  float mu = s / D;
  float var = s2 / D - mu * mu;
  float rs = rsqrtf(var + EPS_);
  float m = HAS_MASK ? mask[row] : 1.f;
#pragma unroll
  for (int e = 0; e < E; ++e) {
    float y = (xv[e] - mu) * rs * g[lane * E + e] + beta[lane * E + e];
    if (HAS_MASK) { y *= m; av[e] *= m; }
    x[base + e] = y;
    acc[base + e] = av[e];
  }
}

// out = x + LN(acc)*g+b
template <int D>
__global__ __launch_bounds__(256) void final_ln_kernel(
    const float* __restrict__ x, const float* __restrict__ acc,
    const float* __restrict__ g, const float* __restrict__ beta,
    float* __restrict__ out, int nrows) {
  constexpr int E = D / 32;
  int lane = threadIdx.x & 31;
  int row = blockIdx.x * (blockDim.x >> 5) + (threadIdx.x >> 5);
  if (row >= nrows) return;
  size_t base = (size_t)row * D + lane * E;
  float av[E];
  float s = 0.f, s2 = 0.f;
#pragma unroll
  for (int e = 0; e < E; ++e) {
    float v = acc[base + e];
    av[e] = v;
    s += v;
    s2 += v * v;
  }
  s = wave_sum32(s);
  s2 = wave_sum32(s2);
  float mu = s / D;
  float var = s2 / D - mu * mu;
  float rs = rsqrtf(var + EPS_);
#pragma unroll
  for (int e = 0; e < E; ++e)
    out[base + e] = x[base + e] + (av[e] - mu) * rs * g[lane * E + e] + beta[lane * E + e];
}

// dz = zp + left[b,i] + right[b, nl[b,i,k]] ; mode0: dst = dz, mode1: dst += dz
__global__ void ft_combine_z_kernel(const float* __restrict__ zp, const float* __restrict__ lft,
                                    const float* __restrict__ rgt, const int* __restrict__ nl,
                                    float* __restrict__ dst, int mode) {
  size_t idx = (size_t)blockIdx.x * blockDim.x + threadIdx.x;
  if (idx >= (size_t)ROWS_Z * DP) return;
  int c = (int)(idx & (DP - 1));
  size_t gr = idx / DP;
  int b = (int)(gr / (NA_ * KNB));
  int rem = (int)(gr % (NA_ * KNB));
  int i = rem / KNB;
  int g = nl[gr];
  float v = zp[idx] + lft[(size_t)(b * NA_ + i) * DP + c] + rgt[(size_t)(b * NA_ + g) * DP + c];
  if (mode) dst[idx] += v; else dst[idx] = v;
}

// ---------------------------------------------------------------------------
extern "C" void kernel_launch(void* const* d_in, const int* in_sizes, int n_in,
                              void* d_out, int out_size, void* d_ws, size_t ws_size,
                              hipStream_t stream) {
  (void)in_sizes; (void)n_in; (void)out_size; (void)ws_size;
  // params in setup_inputs() dict insertion order
  const float* pWatom = (const float*)d_in[0];
  const float* pbatom = (const float*)d_in[1];
  const float* pWpair = (const float*)d_in[2];
  const float* pbpair = (const float*)d_in[3];
  const float* pWleft = (const float*)d_in[4];
  const float* pbleft = (const float*)d_in[5];
  const float* pWright = (const float*)d_in[6];
  const float* pbright = (const float*)d_in[7];
  const float* pWq = (const float*)d_in[8];
  const float* pWk = (const float*)d_in[9];
  const float* pWv = (const float*)d_in[10];
  const float* pWb = (const float*)d_in[11];
  const float* pWo = (const float*)d_in[12];
  const float* pbo = (const float*)d_in[13];
  const float* pnln1g = (const float*)d_in[14];
  const float* pnln1b = (const float*)d_in[15];
  const float* pnln2g = (const float*)d_in[16];
  const float* pnln2b = (const float*)d_in[17];
  const float* pntW1 = (const float*)d_in[18];
  const float* pntb1 = (const float*)d_in[19];
  const float* pntW2 = (const float*)d_in[20];
  const float* pntb2 = (const float*)d_in[21];
  const float* popWa = (const float*)d_in[22];
  const float* popWb = (const float*)d_in[23];
  const float* popWp = (const float*)d_in[24];
  const float* popbp = (const float*)d_in[25];
  const float* peln1g = (const float*)d_in[26];
  const float* peln1b = (const float*)d_in[27];
  const float* peln2g = (const float*)d_in[28];
  const float* peln2b = (const float*)d_in[29];
  const float* petW1 = (const float*)d_in[30];
  const float* petb1 = (const float*)d_in[31];
  const float* petW2 = (const float*)d_in[32];
  const float* petb2 = (const float*)d_in[33];
  const float* pfag = (const float*)d_in[34];
  const float* pfab = (const float*)d_in[35];
  const float* pfpg = (const float*)d_in[36];
  const float* pfpb = (const float*)d_in[37];
  const float* atom_raw = (const float*)d_in[38];
  const float* atom_mask = (const float*)d_in[39];
  const float* pair_raw = (const float*)d_in[40];
  const float* pair_mask = (const float*)d_in[41];
  const int* nl = (const int*)d_in[42];

  // -------- workspace carve-up --------
  char* cur = (char*)d_ws;
  auto alloc = [&](size_t bytes) -> void* {
    void* p = cur;
    cur += (bytes + 255) & ~(size_t)255;
    return p;
  };
  half_t *wtA[4], *wtP[4], *wtLf[4], *wtRg[4];
  for (int i = 0; i < 4; ++i) {
    wtA[i] = (half_t*)alloc((size_t)DA * 160 * 2);
    wtP[i] = (half_t*)alloc((size_t)DP * 32 * 2);
    wtLf[i] = (half_t*)alloc((size_t)DP * 160 * 2);
    wtRg[i] = (half_t*)alloc((size_t)DP * 160 * 2);
  }
  half_t *wtQ[NLAY], *wtK[NLAY], *wtV[NLAY], *wtO[NLAY], *wtN1[NLAY], *wtN2[NLAY];
  half_t *wtOa[NLAY], *wtOb[NLAY], *wtOp[NLAY], *wtE1[NLAY], *wtE2[NLAY];
  for (int l = 0; l < NLAY; ++l) {
    wtQ[l] = (half_t*)alloc((size_t)DA * DA * 2);
    wtK[l] = (half_t*)alloc((size_t)DA * DA * 2);
    wtV[l] = (half_t*)alloc((size_t)DA * DA * 2);
    wtO[l] = (half_t*)alloc((size_t)DA * DA * 2);
    wtN1[l] = (half_t*)alloc((size_t)256 * DA * 2);
    wtN2[l] = (half_t*)alloc((size_t)DA * 256 * 2);
    wtOa[l] = (half_t*)alloc((size_t)DOP * DA * 2);
    wtOb[l] = (half_t*)alloc((size_t)DOP * DA * 2);
    wtOp[l] = (half_t*)alloc((size_t)DP * DOP * 2);
    wtE1[l] = (half_t*)alloc((size_t)128 * DP * 2);
    wtE2[l] = (half_t*)alloc((size_t)DP * 128 * 2);
  }
  float* a = (float*)alloc((size_t)ROWS_A * DA * 4);
  float* acca = (float*)alloc((size_t)ROWS_A * DA * 4);
  float* qp = (float*)alloc((size_t)ROWS_A * DA * 4);
  float* kpb = (float*)alloc((size_t)ROWS_A * DA * 4);
  float* vpb = (float*)alloc((size_t)ROWS_A * DA * 4);
  float* obuf = (float*)alloc((size_t)ROWS_A * DA * 4);
  float* dA = (float*)alloc((size_t)ROWS_A * DA * 4);
  float* lft = (float*)alloc((size_t)ROWS_A * DP * 4);
  float* rgt = (float*)alloc((size_t)ROWS_A * DP * 4);
  float* lb = (float*)alloc((size_t)ROWS_A * DOP * 4);
  float* rp = (float*)alloc((size_t)ROWS_A * DOP * 4);
  float* z = (float*)alloc((size_t)ROWS_Z * DP * 4);
  float* accz = (float*)alloc((size_t)ROWS_Z * DP * 4);
  float* dZ = (float*)alloc((size_t)ROWS_Z * DP * 4);

  // -------- one-time-ish weight conversion (cheap, graph-safe) --------
  auto conv = [&](const float* s, half_t* d, int K, int N, int KP) {
    int n = N * KP;
    convert_wt_kernel<<<(n + 255) / 256, 256, 0, stream>>>(s, d, K, N, KP);
  };
  for (int i = 0; i < 4; ++i) {
    conv(pWatom + (size_t)i * CF * DA, wtA[i], CF, DA, 160);
    conv(pWpair + (size_t)i * CPF * DP, wtP[i], CPF, DP, 32);
    conv(pWleft + (size_t)i * CF * DP, wtLf[i], CF, DP, 160);
    conv(pWright + (size_t)i * CF * DP, wtRg[i], CF, DP, 160);
  }
  for (int l = 0; l < NLAY; ++l) {
    conv(pWq + (size_t)l * DA * DA, wtQ[l], DA, DA, 128);
    conv(pWk + (size_t)l * DA * DA, wtK[l], DA, DA, 128);
    conv(pWv + (size_t)l * DA * DA, wtV[l], DA, DA, 128);
    conv(pWo + (size_t)l * DA * DA, wtO[l], DA, DA, 128);
    conv(pntW1 + (size_t)l * DA * 256, wtN1[l], DA, 256, 128);
    conv(pntW2 + (size_t)l * 256 * DA, wtN2[l], 256, DA, 256);
    conv(popWa + (size_t)l * DA * DOP, wtOa[l], DA, DOP, 128);
    conv(popWb + (size_t)l * DA * DOP, wtOb[l], DA, DOP, 128);
    conv(popWp + (size_t)l * DOP * DP, wtOp[l], DOP, DP, 32);
    conv(petW1 + (size_t)l * DP * 128, wtE1[l], DP, 128, 64);
    conv(petW2 + (size_t)l * 128 * DP, wtE2[l], 128, DP, 128);
  }

  fill0_kernel<<<(unsigned)(((size_t)ROWS_A * DA + 255) / 256), 256, 0, stream>>>(a, (size_t)ROWS_A * DA);
  fill0_kernel<<<(unsigned)(((size_t)ROWS_Z * DP + 255) / 256), 256, 0, stream>>>(z, (size_t)ROWS_Z * DP);

  // FeatureTransformer instance: mode 0 -> acc (ft0), mode 1 -> add into a/z
  auto run_ft = [&](int inst, int mode) {
    gemm_rows_kernel<CF, 160, DA><<<ROWS_A / 64, 128, 0, stream>>>(
        atom_raw, wtA[inst], pbatom + (size_t)inst * DA, mode ? a : acca, mode);
    gemm_rows_kernel<CPF, 32, DP><<<ROWS_Z / 64, 128, 0, stream>>>(
        pair_raw, wtP[inst], pbpair + (size_t)inst * DP, dZ, 0);
    gemm_rows_kernel<CF, 160, DP><<<ROWS_A / 64, 128, 0, stream>>>(
        atom_raw, wtLf[inst], pbleft + (size_t)inst * DP, lft, 0);
    gemm_rows_kernel<CF, 160, DP><<<ROWS_A / 64, 128, 0, stream>>>(
        atom_raw, wtRg[inst], pbright + (size_t)inst * DP, rgt, 0);
    ft_combine_z_kernel<<<(unsigned)(((size_t)ROWS_Z * DP + 255) / 256), 256, 0, stream>>>(
        dZ, lft, rgt, nl, mode ? z : accz, mode);
  };

  run_ft(0, 0);

  for (int l = 0; l < NLAY; ++l) {
    run_ft(l + 1, 1);
    // Q/K/V projections on a (gather commutes with the linear maps)
    gemm_rows_kernel<DA, 128, DA><<<ROWS_A / 64, 128, 0, stream>>>(a, wtQ[l], nullptr, qp, 0);
    gemm_rows_kernel<DA, 128, DA><<<ROWS_A / 64, 128, 0, stream>>>(a, wtK[l], nullptr, kpb, 0);
    gemm_rows_kernel<DA, 128, DA><<<ROWS_A / 64, 128, 0, stream>>>(a, wtV[l], nullptr, vpb, 0);
    attention_kernel<<<ROWS_A, 128, 0, stream>>>(qp, kpb, vpb, z, pWb + (size_t)l * DP * NH,
                                                 atom_mask, nl, obuf);
    gemm_rows_kernel<DA, 128, DA><<<ROWS_A / 64, 128, 0, stream>>>(
        obuf, wtO[l], pbo + (size_t)l * DA, dA, 0);
    residual_ln_kernel<DA, false><<<ROWS_A / 8, 256, 0, stream>>>(
        a, dA, acca, pnln1g + (size_t)l * DA, pnln1b + (size_t)l * DA, nullptr, ROWS_A);
    mlp_kernel<DA, 128, 256, DA><<<ROWS_A / 64, 128, 0, stream>>>(
        a, wtN1[l], pntb1 + (size_t)l * 256, wtN2[l], pntb2 + (size_t)l * DA, dA);
    residual_ln_kernel<DA, true><<<ROWS_A / 8, 256, 0, stream>>>(
        a, dA, acca, pnln2g + (size_t)l * DA, pnln2b + (size_t)l * DA, atom_mask, ROWS_A);
    // pair update
    gemm_rows_kernel<DA, 128, DOP><<<ROWS_A / 64, 128, 0, stream>>>(a, wtOa[l], nullptr, lb, 0);
    gemm_rows_kernel<DA, 128, DOP><<<ROWS_A / 64, 128, 0, stream>>>(a, wtOb[l], nullptr, rp, 0);
    outer_product_kernel<<<ROWS_Z / 64, 128, 0, stream>>>(
        lb, rp, atom_mask, nl, wtOp[l], popbp + (size_t)l * DP, dZ);
    residual_ln_kernel<DP, false><<<ROWS_Z / 8, 256, 0, stream>>>(
        z, dZ, accz, peln1g + (size_t)l * DP, peln1b + (size_t)l * DP, nullptr, ROWS_Z);
    mlp_kernel<DP, 64, 128, DP><<<ROWS_Z / 64, 128, 0, stream>>>(
        z, wtE1[l], petb1 + (size_t)l * 128, wtE2[l], petb2 + (size_t)l * DP, dZ);
    residual_ln_kernel<DP, true><<<ROWS_Z / 8, 256, 0, stream>>>(
        z, dZ, accz, peln2g + (size_t)l * DP, peln2b + (size_t)l * DP, pair_mask, ROWS_Z);
  }

  float* out = (float*)d_out;
  final_ln_kernel<DA><<<ROWS_A / 8, 256, 0, stream>>>(a, acca, pfag, pfab, out, ROWS_A);
  final_ln_kernel<DP><<<ROWS_Z / 8, 256, 0, stream>>>(z, accz, pfpg, pfpb,
                                                      out + (size_t)ROWS_A * DA, ROWS_Z);
}
